// TwoCompLayer_9990093930915
// MI455X (gfx1250) — matrix operand order repaired and checked by reading, not compiled, not run on hardware
//
#include <hip/hip_runtime.h>

// ---------------- problem constants ----------------
constexpr int   kB    = 8192;
constexpr int   kNIn  = 1024;
constexpr int   kNOut = 1024;
constexpr float kTau   = 0.8f;
constexpr float kAlpha = 0.05f;
constexpr float kHebb  = 1e-5f;

// ---------------- types ----------------
typedef __bf16 bhalf;
typedef __attribute__((ext_vector_type(16))) bhalf        v16bf;
typedef __attribute__((ext_vector_type(8)))  float        v8f;
typedef __attribute__((ext_vector_type(4)))  unsigned int u32x4;

__device__ __forceinline__ unsigned short f2bf(float f) {
  unsigned u = __float_as_uint(f);
  u += 0x7FFFu + ((u >> 16) & 1u);   // round-to-nearest-even
  return (unsigned short)(u >> 16);
}

// ---------------- f32 -> bf16 flat convert (4 elems/thread) ----------------
__global__ void cvt_bf16_kernel(const float* __restrict__ in,
                                unsigned short* __restrict__ out, int n4) {
  int i = blockIdx.x * blockDim.x + threadIdx.x;
  if (i < n4) {
    const float4 v = reinterpret_cast<const float4*>(in)[i];
    ushort4 o;
    o.x = f2bf(v.x); o.y = f2bf(v.y); o.z = f2bf(v.z); o.w = f2bf(v.w);
    reinterpret_cast<ushort4*>(out)[i] = o;
  }
}

// ------------- f32 [rows,cols] -> bf16 transposed [cols,rows] -------------
__global__ void trans_bf16_kernel(const float* __restrict__ in,
                                  unsigned short* __restrict__ out,
                                  int rows, int cols) {
  __shared__ unsigned short tile[32][33];
  int j0 = blockIdx.x * 32, i0 = blockIdx.y * 32;
#pragma unroll
  for (int r = 0; r < 4; ++r) {
    int i = i0 + threadIdx.y + r * 8;
    tile[threadIdx.y + r * 8][threadIdx.x] =
        f2bf(in[(size_t)i * cols + j0 + threadIdx.x]);
  }
  __syncthreads();
#pragma unroll
  for (int r = 0; r < 4; ++r) {
    int j = j0 + threadIdx.y + r * 8;
    out[(size_t)j * rows + i0 + threadIdx.x] = tile[threadIdx.x][threadIdx.y + r * 8];
  }
}

// ------------- soma: v_s' , spikes, ebar' (+ transposed bf16 ebar') -------------
__global__ void soma_kernel(const float* __restrict__ vb, const float* __restrict__ va,
                            const float* __restrict__ vs_in, const float* __restrict__ eb_in,
                            float* __restrict__ vs_out, float* __restrict__ spk,
                            float* __restrict__ eb_out, unsigned short* __restrict__ ebT) {
  __shared__ unsigned short tile[32][33];
  int j0 = blockIdx.x * 32, i0 = blockIdx.y * 32;
#pragma unroll
  for (int r = 0; r < 4; ++r) {
    int i = i0 + threadIdx.y + r * 8;
    size_t idx = (size_t)i * kNOut + j0 + threadIdx.x;
    float b = vb[idx], a = va[idx];
    float s = kTau * vs_in[idx] + (1.0f - kTau) * (b - a);
    vs_out[idx] = s;
    spk[idx]    = (s > 0.0f) ? 1.0f : 0.0f;
    float e = (1.0f - kAlpha) * eb_in[idx] + kAlpha * (s - b);
    eb_out[idx] = e;
    tile[threadIdx.y + r * 8][threadIdx.x] = f2bf(e);
  }
  __syncthreads();
#pragma unroll
  for (int r = 0; r < 4; ++r) {
    int j = j0 + threadIdx.y + r * 8;
    ebT[(size_t)j * kB + i0 + threadIdx.x] = tile[threadIdx.x][threadIdx.y + r * 8];
  }
}

// ------------- trace: tr' = (1-a)*tr + a*sig (+ transposed bf16 copy) -------------
__global__ void trace_kernel(const float* __restrict__ sig, const float* __restrict__ tr_in,
                             float* __restrict__ tr_out, unsigned short* __restrict__ trT,
                             int cols) {
  __shared__ unsigned short tile[32][33];
  int j0 = blockIdx.x * 32, i0 = blockIdx.y * 32;
#pragma unroll
  for (int r = 0; r < 4; ++r) {
    int i = i0 + threadIdx.y + r * 8;
    size_t idx = (size_t)i * cols + j0 + threadIdx.x;
    float t = (1.0f - kAlpha) * tr_in[idx] + kAlpha * sig[idx];
    tr_out[idx] = t;
    tile[threadIdx.y + r * 8][threadIdx.x] = f2bf(t);
  }
  __syncthreads();
#pragma unroll
  for (int r = 0; r < 4; ++r) {
    int j = j0 + threadIdx.y + r * 8;
    trT[(size_t)j * kB + i0 + threadIdx.x] = tile[threadIdx.x][threadIdx.y + r * 8];
  }
}

// ---------------- bf16 WMMA GEMM: Out = sa*(A @ Bt^T) + sc*Cin ----------------
// A : [M,K] bf16 row-major   Bt : [N,K] bf16 row-major (B transposed)
// Block tile 128x128, 8 waves (2x4), wave tile 64x32 (8 WMMA accumulators).
// K staged 32 at a time into double-buffered LDS with CDNA5 async global->LDS
// copies (ASYNCcnt), so tile k+1's DMA overlaps tile k's WMMAs.
constexpr int LSTR = 56;  // LDS row stride (bf16): 112B, 16B aligned, bank-spread

__global__ __launch_bounds__(256) void
gemm_bf16_kernel(const unsigned short* __restrict__ A,
                 const unsigned short* __restrict__ Bt,
                 const float* __restrict__ Cin, float* __restrict__ Out,
                 int M, int N, int K, float sa, float sc) {
  __shared__ alignas(16) unsigned short As[2][128 * LSTR];
  __shared__ alignas(16) unsigned short Bs[2][128 * LSTR];

  const int tid  = threadIdx.x;
  const int lane = tid & 31;
  const int wave = tid >> 5;
  const int wm   = wave >> 2;   // 0..1  (M)
  const int wn   = wave & 3;    // 0..3  (N)
  const int m0   = blockIdx.y * 128;
  const int n0   = blockIdx.x * 128;
  const int lm   = lane & 15;
  const int lhi  = lane >> 4;

  v8f zero = {0.f, 0.f, 0.f, 0.f, 0.f, 0.f, 0.f, 0.f};
  v8f acc[4][2];
#pragma unroll
  for (int s = 0; s < 4; ++s)
#pragma unroll
    for (int t = 0; t < 2; ++t) acc[s][t] = zero;

  union Frag { u32x4 u[2]; v16bf v; };

  // async-stage one 128x32 bf16 tile of A and Bt into LDS buffer `buf`
  auto stage = [&](int buf, int kk) {
#pragma unroll
    for (int h = 0; h < 2; ++h) {
      int q = tid + h * 256;
      int r = q >> 2;
      int c = (q & 3) * 8;
      unsigned la = (unsigned)(size_t)(&As[buf][r * LSTR + c]);   // LDS byte offset
      unsigned long long ga =
          (unsigned long long)(A + (size_t)(m0 + r) * K + kk + c);
      asm volatile("global_load_async_to_lds_b128 %0, %1, off"
                   :: "v"(la), "v"(ga) : "memory");
      unsigned lb = (unsigned)(size_t)(&Bs[buf][r * LSTR + c]);
      unsigned long long gb =
          (unsigned long long)(Bt + (size_t)(n0 + r) * K + kk + c);
      asm volatile("global_load_async_to_lds_b128 %0, %1, off"
                   :: "v"(lb), "v"(gb) : "memory");
    }
  };

  stage(0, 0);
  asm volatile("s_wait_asynccnt 0x0" ::: "memory");
  __syncthreads();

  int cur = 0;
  for (int kk = 0; kk < K; kk += 32) {
    if (kk + 32 < K) stage(cur ^ 1, kk + 32);   // DMA next tile under compute

    // A frag (16x32): lanes 0-15 -> K 0..7 & 16..23, lanes 16-31 -> K 8..15 & 24..31
    Frag a[4], b[2];
    const int kbA = lhi * 8;
    const int koB = lhi * 8;
#pragma unroll
    for (int s = 0; s < 4; ++s) {
      int row = (wm * 64 + s * 16 + lm) * LSTR;
      a[s].u[0] = *reinterpret_cast<const u32x4*>(&As[cur][row + kbA]);
      a[s].u[1] = *reinterpret_cast<const u32x4*>(&As[cur][row + 16 + kbA]);
    }
    // B frag (32x16): lanes 0-15 -> K 0..7 & 16..23, lanes 16-31 -> K 8..15 & 24..31 (N = lane%16)
#pragma unroll
    for (int t = 0; t < 2; ++t) {
      int row = (wn * 32 + t * 16 + lm) * LSTR;
      b[t].u[0] = *reinterpret_cast<const u32x4*>(&Bs[cur][row + koB]);
      b[t].u[1] = *reinterpret_cast<const u32x4*>(&Bs[cur][row + 16 + koB]);
    }
#pragma unroll
    for (int s = 0; s < 4; ++s)
#pragma unroll
      for (int t = 0; t < 2; ++t)
        acc[s][t] = __builtin_amdgcn_wmma_f32_16x16x32_bf16(
            false, a[s].v, false, b[t].v, (short)0, acc[s][t], false, false);

    asm volatile("s_wait_asynccnt 0x0" ::: "memory");  // own async writes done
    __syncthreads();                                   // visible to all waves
    cur ^= 1;
  }

  // epilogue: C/D layout -> VGPR r: M = base + r (+8 for lanes 16-31), N = lane%16
#pragma unroll
  for (int s = 0; s < 4; ++s)
#pragma unroll
    for (int t = 0; t < 2; ++t) {
      int gm = m0 + wm * 64 + s * 16 + lhi * 8;
      int gn = n0 + wn * 32 + t * 16 + lm;
#pragma unroll
      for (int r = 0; r < 8; ++r) {
        size_t idx = (size_t)(gm + r) * N + gn;
        Out[idx] = sa * acc[s][t][r] + sc * Cin[idx];
      }
    }
}

// ---------------- launch ----------------
extern "C" void kernel_launch(void* const* d_in, const int* in_sizes, int n_in,
                              void* d_out, int out_size, void* d_ws, size_t ws_size,
                              hipStream_t stream) {
  const float* x    = (const float*)d_in[0];
  const float* fb   = (const float*)d_in[1];
  const float* Wff  = (const float*)d_in[2];
  const float* Wfb  = (const float*)d_in[3];
  const float* vb   = (const float*)d_in[4];
  const float* va   = (const float*)d_in[5];
  const float* vs   = (const float*)d_in[6];
  const float* rbar = (const float*)d_in[7];
  const float* ebar = (const float*)d_in[8];

  float* out    = (float*)d_out;
  float* o_spk  = out;
  float* o_wff  = o_spk  + (size_t)kB * kNOut;
  float* o_vb   = o_wff  + (size_t)kNIn * kNOut;
  float* o_va   = o_vb   + (size_t)kB * kNOut;
  float* o_vs   = o_va   + (size_t)kB * kNOut;
  float* o_rbar = o_vs   + (size_t)kB * kNOut;
  float* o_ebar = o_rbar + (size_t)kB * kNIn;

  char* ws = (char*)d_ws;
  unsigned short* xb    = (unsigned short*)(ws);             // 16 MB
  unsigned short* fbb   = (unsigned short*)(ws + 16777216);  // 16 MB
  unsigned short* wffT  = (unsigned short*)(ws + 33554432);  //  2 MB
  unsigned short* wfbT  = (unsigned short*)(ws + 35651584);  //  2 MB
  unsigned short* rbarT = xb;    // reused after GEMM1 consumed xb
  unsigned short* ebarT = fbb;   // reused after GEMM2 consumed fbb

  dim3 blk256(256);
  dim3 tb(32, 8);

  // 1) precision conversion
  cvt_bf16_kernel<<<dim3((kB * kNIn / 4) / 256), blk256, 0, stream>>>(x, xb, kB * kNIn / 4);
  cvt_bf16_kernel<<<dim3((kB * kNOut / 4) / 256), blk256, 0, stream>>>(fb, fbb, kB * kNOut / 4);
  trans_bf16_kernel<<<dim3(kNOut / 32, kNIn / 32), tb, 0, stream>>>(Wff, wffT, kNIn, kNOut);
  trans_bf16_kernel<<<dim3(kNOut / 32, kNOut / 32), tb, 0, stream>>>(Wfb, wfbT, kNOut, kNOut);

  // 2) v_b' = 0.8 v_b + 0.2 (x @ Wff);  v_a' = 0.8 v_a + 0.2 (fb @ Wfb)
  gemm_bf16_kernel<<<dim3(kNOut / 128, kB / 128), blk256, 0, stream>>>(
      xb, wffT, vb, o_vb, kB, kNOut, kNIn, 1.0f - kTau, kTau);
  gemm_bf16_kernel<<<dim3(kNOut / 128, kB / 128), blk256, 0, stream>>>(
      fbb, wfbT, va, o_va, kB, kNOut, kNOut, 1.0f - kTau, kTau);

  // 3) soma voltage, spikes, ebar trace (+ transposed bf16 ebar)
  soma_kernel<<<dim3(kNOut / 32, kB / 32), tb, 0, stream>>>(
      o_vb, o_va, vs, ebar, o_vs, o_spk, o_ebar, ebarT);

  // 4) rbar trace (+ transposed bf16 rbar)
  trace_kernel<<<dim3(kNIn / 32, kB / 32), tb, 0, stream>>>(x, rbar, o_rbar, rbarT, kNIn);

  // 5) W_ff' = W_ff + (lr/B) * (rbar'^T @ ebar')
  gemm_bf16_kernel<<<dim3(kNOut / 128, kNIn / 128), blk256, 0, stream>>>(
      rbarT, ebarT, Wff, o_wff, kNIn, kNOut, kB, kHebb / (float)kB, 1.0f);
}
